// MixtureOfExperts_5368709120667
// MI455X (gfx1250) — matrix operand order repaired and checked by reading, not compiled, not run on hardware
//
#include <hip/hip_runtime.h>
#include <stdint.h>

// MoE (8 experts, top-2) for MI455X gfx1250.
// Memory-bound -> pre-cast x/w1/w2 to bf16 once (halves streamed bytes, kills
// per-tile cvt VALU), then dense per-expert grouped GEMMs in bf16 WMMA
// (16x16x32, f32 acc). Routing weight folded into GEMM1 epilogue so GEMM2
// just sums across experts (stream-serialized coalesced RMW of out). A tiles
// fetched with gfx1250 GLOBAL_LOAD_ASYNC_TO_LDS_B128 (ASYNCcnt fence).
// Workspace: combine 512KB | hbuf 32MB | xbf 32MB | w1bf 64MB | w2bf 64MB.

typedef __attribute__((ext_vector_type(16))) __bf16    bf16x16;
typedef __attribute__((ext_vector_type(8)))  float     f32x8;
typedef __attribute__((ext_vector_type(4)))  float     f32x4;
typedef __attribute__((ext_vector_type(4)))  uint32_t  u32x4;
typedef __attribute__((ext_vector_type(2)))  uint32_t  u32x2;

namespace {
constexpr int kE = 8, kH = 1024, kI = 4096, kT = 16384;
constexpr int kChunk = 4096;                 // tokens per chunk
constexpr int BM = 128, BN = 128, BK = 32;   // block tile
constexpr int LDA = BK + 8;                  // padded LDS strides (80B rows)
constexpr int LDB = BK + 8;
constexpr int LDC = BN + 8;                  // bf16 stage stride (272B rows)
constexpr int LDC2 = BN + 4;                 // f32 stage stride (528B rows)
}

__device__ __forceinline__ uint32_t f2bf(float f) {
  uint32_t u = __float_as_uint(f);
  u += 0x7fffu + ((u >> 16) & 1u);           // round-to-nearest-even
  return u >> 16;
}

__device__ __forceinline__ float gelu_exact(float x) {
  return 0.5f * x * (1.0f + erff(x * 0.70710678118654752f));
}

union FragAB { u32x4 q[2]; bf16x16 v; };

// ---------------- fp32 -> bf16 pre-cast (one pass, bandwidth bound) -------
__global__ __launch_bounds__(256)
void cast_f32_bf16(const float* __restrict__ src, uint16_t* __restrict__ dst) {
  const size_t i = ((size_t)blockIdx.x * 256 + threadIdx.x) * 8;
  f32x4 a = *(const f32x4*)(src + i);
  f32x4 b = *(const f32x4*)(src + i + 4);
  u32x4 o;
  o.x = f2bf(a.x) | (f2bf(a.y) << 16);
  o.y = f2bf(a.z) | (f2bf(a.w) << 16);
  o.z = f2bf(b.x) | (f2bf(b.y) << 16);
  o.w = f2bf(b.z) | (f2bf(b.w) << 16);
  *(u32x4*)(dst + i) = o;
}

// ---------------- Router: logits -> softmax -> top2 -> renorm -------------
__global__ __launch_bounds__(256)
void moe_router(const float* __restrict__ x, const float* __restrict__ rw,
                const float* __restrict__ rb, float* __restrict__ comb) {
  const int lane = threadIdx.x & 31;
  const int wid  = threadIdx.x >> 5;
  const int t = blockIdx.x * 8 + wid;        // one wave per token
  const float* xr = x + (size_t)t * kH;
  float acc[8];
#pragma unroll
  for (int e = 0; e < 8; ++e) acc[e] = 0.f;
  for (int j = lane; j < kH; j += 32) {
    float xv = xr[j];
    f32x4 w0 = *(const f32x4*)(rw + (size_t)j * 8);
    f32x4 w1 = *(const f32x4*)(rw + (size_t)j * 8 + 4);
    acc[0] += xv * w0.x; acc[1] += xv * w0.y; acc[2] += xv * w0.z; acc[3] += xv * w0.w;
    acc[4] += xv * w1.x; acc[5] += xv * w1.y; acc[6] += xv * w1.z; acc[7] += xv * w1.w;
  }
#pragma unroll
  for (int off = 16; off > 0; off >>= 1) {
#pragma unroll
    for (int e = 0; e < 8; ++e) acc[e] += __shfl_xor(acc[e], off, 32);
  }
  if (lane == 0) {
    float mx = -1e30f;
#pragma unroll
    for (int e = 0; e < 8; ++e) { acc[e] += rb[e]; mx = fmaxf(mx, acc[e]); }
    float p[8];
#pragma unroll
    for (int e = 0; e < 8; ++e) p[e] = __expf(acc[e] - mx);
    int i1 = 0;
#pragma unroll
    for (int e = 1; e < 8; ++e) if (p[e] > p[i1]) i1 = e;
    int i2 = (i1 == 0) ? 1 : 0;
#pragma unroll
    for (int e = 0; e < 8; ++e) if (e != i1 && p[e] > p[i2]) i2 = e;
    float s = 1.0f / (p[i1] + p[i2]);        // softmax denom cancels in renorm
#pragma unroll
    for (int e = 0; e < 8; ++e)
      comb[(size_t)t * 8 + e] = (e == i1) ? p[i1] * s : (e == i2) ? p[i2] * s : 0.f;
  }
}

// ---------------- GEMM1: h = gelu(Xbf @ W1bf + b1e) * combine (bf16) ------
__global__ __launch_bounds__(256)
void moe_gemm1(const uint16_t* __restrict__ Xb, const uint16_t* __restrict__ W1,
               const float* __restrict__ B1, const float* __restrict__ comb,
               uint16_t* __restrict__ hbuf, int tokBase) {
  __shared__ __align__(16) union {
    struct { uint16_t As[BM * LDA]; uint16_t Bs[BN * LDB]; } t;
    uint16_t stage[BM * LDC];                // C tile staging (34KB)
  } sm;
  const int tid = threadIdx.x;
  const int lane = tid & 31;
  const int wid = tid >> 5;
  const int wm = wid & 1;                    // 2x4 wave grid -> 64x32 wave tile
  const int wn = wid >> 1;
  const int m0 = blockIdx.y * BM;
  const int n0 = blockIdx.x * BN;
  const uint32_t asBase = (uint32_t)(uintptr_t)(&sm.t.As[0]);  // LDS byte offset

  f32x8 acc[4][2] = {};

  for (int k0 = 0; k0 < kH; k0 += BK) {
#pragma unroll
    for (int q = 0; q < 2; ++q) {            // A (bf16): async global -> LDS, b128
      int flat = q * 256 + tid;
      int row = flat >> 2, cg = (flat & 3) << 3;
      uint32_t ldsa = asBase + (uint32_t)(row * LDA + cg) * 2u;
      const uint16_t* gp = Xb + (size_t)(tokBase + m0 + row) * kH + (k0 + cg);
      asm volatile("global_load_async_to_lds_b128 %0, %1, off"
                   :: "v"(ldsa), "v"(gp) : "memory");
    }
#pragma unroll
    for (int p = 0; p < 4; ++p) {            // B: 32x128 bf16 -> transposed LDS
      int flat = p * 256 + tid;
      int kr = flat >> 5, ng = (flat & 31) << 2;
      const uint16_t* gp = W1 + (size_t)(k0 + kr) * kI + (n0 + ng);
      u32x2 v = *(const u32x2*)gp;
      if (p == 0 && k0 + BK < kH)
        __builtin_prefetch(gp + (size_t)BK * kI, 0, 1);   // global_prefetch
      sm.t.Bs[(ng + 0) * LDB + kr] = (uint16_t)(v.x);
      sm.t.Bs[(ng + 1) * LDB + kr] = (uint16_t)(v.x >> 16);
      sm.t.Bs[(ng + 2) * LDB + kr] = (uint16_t)(v.y);
      sm.t.Bs[(ng + 3) * LDB + kr] = (uint16_t)(v.y >> 16);
    }
    asm volatile("s_wait_asynccnt 0" ::: "memory");
    __syncthreads();
    FragAB a[4], b[2];
#pragma unroll
    for (int mi = 0; mi < 4; ++mi) {         // A frag: lane=row, K halves @0/8 (+16)
      int row = wm * 64 + mi * 16 + (lane & 15);
      int kb = (lane < 16) ? 0 : 8;
      const uint16_t* ptr = &sm.t.As[row * LDA + kb];
      a[mi].q[0] = *(const u32x4*)(ptr);
      a[mi].q[1] = *(const u32x4*)(ptr + 16);
    }
#pragma unroll
    for (int ni = 0; ni < 2; ++ni) {         // B frag: lane=col, K halves @0/8 (+16)
      int col = wn * 32 + ni * 16 + (lane & 15);
      int kb = (lane < 16) ? 0 : 8;
      const uint16_t* ptr = &sm.t.Bs[col * LDB + kb];
      b[ni].q[0] = *(const u32x4*)(ptr);
      b[ni].q[1] = *(const u32x4*)(ptr + 16);
    }
#pragma unroll
    for (int mi = 0; mi < 4; ++mi)
#pragma unroll
      for (int ni = 0; ni < 2; ++ni)
        acc[mi][ni] = __builtin_amdgcn_wmma_f32_16x16x32_bf16(
            false, a[mi].v, false, b[ni].v, (short)0, acc[mi][ni], false, false);
    __syncthreads();
  }

  // Epilogue: bias + gelu + combine, stage bf16 tile in LDS, coalesced store.
  {
    const int rhalf = (lane >> 4) * 8;
    const int nl = lane & 15;
#pragma unroll
    for (int mi = 0; mi < 4; ++mi) {
#pragma unroll
      for (int ni = 0; ni < 2; ++ni) {
        int nloc = wn * 32 + ni * 16 + nl;
        float bias = B1[n0 + nloc];
#pragma unroll
        for (int r = 0; r < 8; ++r) {
          int mloc = wm * 64 + mi * 16 + rhalf + r;
          float c = comb[(size_t)(tokBase + m0 + mloc) * 8];  // pre-offset by expert
          sm.stage[mloc * LDC + nloc] =
              (uint16_t)f2bf(gelu_exact(acc[mi][ni][r] + bias) * c);
        }
      }
    }
  }
  __syncthreads();
#pragma unroll
  for (int q = 0; q < 8; ++q) {              // 128 rows x 256B, b128 bursts
    int flat = q * 256 + tid;
    int row = flat >> 4, cg = (flat & 15) << 3;
    u32x4 v = *(const u32x4*)&sm.stage[row * LDC + cg];
    *(u32x4*)(hbuf + (size_t)(m0 + row) * kI + (n0 + cg)) = v;
  }
}

// ---------------- GEMM2: out (+)= h @ W2bf + combine*b2e ------------------
__global__ __launch_bounds__(256)
void moe_gemm2(const uint16_t* __restrict__ Hb, const uint16_t* __restrict__ W2,
               const float* __restrict__ B2, const float* __restrict__ comb,
               float* __restrict__ out, int tokBase, int accumulate) {
  __shared__ __align__(16) union {
    struct { uint16_t As[BM * LDA]; uint16_t Bs[BN * LDB]; } t;
    float stage[BM * LDC2];                  // f32 C tile staging (66KB)
  } sm;
  const int tid = threadIdx.x;
  const int lane = tid & 31;
  const int wid = tid >> 5;
  const int wm = wid & 1;
  const int wn = wid >> 1;
  const int m0 = blockIdx.y * BM;
  const int n0 = blockIdx.x * BN;
  const uint32_t asBase = (uint32_t)(uintptr_t)(&sm.t.As[0]);

  f32x8 acc[4][2] = {};

  for (int k0 = 0; k0 < kI; k0 += BK) {
#pragma unroll
    for (int q = 0; q < 2; ++q) {            // A (bf16): async global -> LDS, b128
      int flat = q * 256 + tid;
      int row = flat >> 2, cg = (flat & 3) << 3;
      uint32_t ldsa = asBase + (uint32_t)(row * LDA + cg) * 2u;
      const uint16_t* gp = Hb + (size_t)(m0 + row) * kI + (k0 + cg);
      asm volatile("global_load_async_to_lds_b128 %0, %1, off"
                   :: "v"(ldsa), "v"(gp) : "memory");
    }
#pragma unroll
    for (int p = 0; p < 4; ++p) {            // W2: 32x128 bf16 -> transposed LDS
      int flat = p * 256 + tid;
      int kr = flat >> 5, ng = (flat & 31) << 2;
      const uint16_t* gp = W2 + (size_t)(k0 + kr) * kH + (n0 + ng);
      u32x2 v = *(const u32x2*)gp;
      if (p == 0 && k0 + BK < kI)
        __builtin_prefetch(gp + (size_t)BK * kH, 0, 1);
      sm.t.Bs[(ng + 0) * LDB + kr] = (uint16_t)(v.x);
      sm.t.Bs[(ng + 1) * LDB + kr] = (uint16_t)(v.x >> 16);
      sm.t.Bs[(ng + 2) * LDB + kr] = (uint16_t)(v.y);
      sm.t.Bs[(ng + 3) * LDB + kr] = (uint16_t)(v.y >> 16);
    }
    asm volatile("s_wait_asynccnt 0" ::: "memory");
    __syncthreads();
    FragAB a[4], b[2];
#pragma unroll
    for (int mi = 0; mi < 4; ++mi) {
      int row = wm * 64 + mi * 16 + (lane & 15);
      int kb = (lane < 16) ? 0 : 8;
      const uint16_t* ptr = &sm.t.As[row * LDA + kb];
      a[mi].q[0] = *(const u32x4*)(ptr);
      a[mi].q[1] = *(const u32x4*)(ptr + 16);
    }
#pragma unroll
    for (int ni = 0; ni < 2; ++ni) {
      int col = wn * 32 + ni * 16 + (lane & 15);
      int kb = (lane < 16) ? 0 : 8;
      const uint16_t* ptr = &sm.t.Bs[col * LDB + kb];
      b[ni].q[0] = *(const u32x4*)(ptr);
      b[ni].q[1] = *(const u32x4*)(ptr + 16);
    }
#pragma unroll
    for (int mi = 0; mi < 4; ++mi)
#pragma unroll
      for (int ni = 0; ni < 2; ++ni)
        acc[mi][ni] = __builtin_amdgcn_wmma_f32_16x16x32_bf16(
            false, a[mi].v, false, b[ni].v, (short)0, acc[mi][ni], false, false);
    __syncthreads();
  }

  // Epilogue: +combine*b2, stage f32 tile in LDS, coalesced b128 RMW of out.
  {
    const int rhalf = (lane >> 4) * 8;
    const int nl = lane & 15;
#pragma unroll
    for (int mi = 0; mi < 4; ++mi) {
#pragma unroll
      for (int ni = 0; ni < 2; ++ni) {
        int nloc = wn * 32 + ni * 16 + nl;
        float b2v = B2[n0 + nloc];
#pragma unroll
        for (int r = 0; r < 8; ++r) {
          int mloc = wm * 64 + mi * 16 + rhalf + r;
          float c = comb[(size_t)(tokBase + m0 + mloc) * 8];
          sm.stage[mloc * LDC2 + nloc] = acc[mi][ni][r] + c * b2v;
        }
      }
    }
  }
  __syncthreads();
#pragma unroll
  for (int q = 0; q < 16; ++q) {             // 128 rows x 512B, b128 RMW bursts
    int flat = q * 256 + tid;
    int row = flat >> 5, cg = (flat & 31) << 2;
    f32x4 v = *(const f32x4*)&sm.stage[row * LDC2 + cg];
    float* op = out + (size_t)(tokBase + m0 + row) * kH + (n0 + cg);
    if (accumulate) { f32x4 o = *(const f32x4*)op; v = v + o; }
    *(f32x4*)op = v;
  }
}

extern "C" void kernel_launch(void* const* d_in, const int* in_sizes, int n_in,
                              void* d_out, int out_size, void* d_ws, size_t ws_size,
                              hipStream_t stream) {
  const float* x  = (const float*)d_in[0];   // [T,H]
  const float* w1 = (const float*)d_in[1];   // [E,H,I]
  const float* b1 = (const float*)d_in[2];   // [E,I]
  const float* w2 = (const float*)d_in[3];   // [E,I,H]
  const float* b2 = (const float*)d_in[4];   // [E,H]
  const float* rw = (const float*)d_in[5];   // [H,E]
  const float* rb = (const float*)d_in[6];   // [E]
  float* out = (float*)d_out;

  // Workspace layout (~193MB).
  char* wsb = (char*)d_ws;
  float*    comb = (float*)wsb;                                  // 512KB
  uint16_t* hbuf = (uint16_t*)(wsb + (size_t)kT * kE * 4);       // 32MB
  uint16_t* xbf  = hbuf + (size_t)kChunk * kI;                   // 32MB
  uint16_t* w1bf = xbf + (size_t)kT * kH;                        // 64MB
  uint16_t* w2bf = w1bf + (size_t)kE * kH * kI;                  // 64MB

  // One-time bf16 pre-cast of the streamed operands.
  cast_f32_bf16<<<(int)(((size_t)kT * kH) / 2048), 256, 0, stream>>>(x, xbf);
  cast_f32_bf16<<<(int)(((size_t)kE * kH * kI) / 2048), 256, 0, stream>>>(w1, w1bf);
  cast_f32_bf16<<<(int)(((size_t)kE * kI * kH) / 2048), 256, 0, stream>>>(w2, w2bf);

  moe_router<<<kT / 8, 256, 0, stream>>>(x, rw, rb, comb);

  for (int c = 0; c < kT / kChunk; ++c) {
    const int tokBase = c * kChunk;
    for (int e = 0; e < kE; ++e) {
      moe_gemm1<<<dim3(kI / BN, kChunk / BM), 256, 0, stream>>>(
          xbf, w1bf + (size_t)e * kH * kI, b1 + (size_t)e * kI, comb + e, hbuf,
          tokBase);
      moe_gemm2<<<dim3(kH / BN, kChunk / BM), 256, 0, stream>>>(
          hbuf, w2bf + (size_t)e * kI * kH, b2 + (size_t)e * kH, comb + e, out,
          tokBase, e > 0 ? 1 : 0);
    }
  }
}
